// TSM2_5789615915062
// MI455X (gfx1250) — matrix operand order repaired and checked by reading, not compiled, not run on hardware
//
#include <hip/hip_runtime.h>
#include <math.h>

// ---------------- model dims ----------------
#define B_    16
#define L_    512
#define M_    32
#define PRED_ 96
#define P_    16
#define S_    8
#define D_    128
#define NST_  16
#define DC_   4
#define NL_   4
#define N_    63          // (L-P)/S + 1
#define DI_   256         // EXP*D
#define DTR_  8           // dt_rank
#define BM_   512         // B*M
#define XP_   40          // DTR + 2*NST
#define TTOK_ 32256       // BM_*N_ == (B_*N_)*M_

// ---------------- WMMA types ----------------
typedef __bf16 v16bf __attribute__((ext_vector_type(16)));
typedef float  v8f   __attribute__((ext_vector_type(8)));
typedef float  v4f   __attribute__((ext_vector_type(4)));
typedef __attribute__((address_space(3))) float lds_f;   // true LDS pointer (32-bit offset in VGPR)

union BF16Frag { v16bf v; __bf16 e[16]; };

__device__ __forceinline__ float siluf(float x) { return x / (1.f + __expf(-x)); }
__device__ __forceinline__ float softplusf(float x) { return x > 20.f ? x : log1pf(__expf(x)); }

// =====================================================================
// GEMM: C[T x Nout] = A[T x K] @ W[Nout x K]^T (+bias), bf16 WMMA.
//
// Block = 256 threads = 8 waves. Block computes a 128-row x 64-col tile:
//   wave w -> rows [tmBlock*128 + w*16, +16), all waves share one 64-col strip.
// B strip (64 cols x 32 K f32 = 8KB) is staged global->LDS with
// global_load_async_to_lds_b128 (ASYNCcnt), double buffered (16KB LDS),
// then each wave issues 4 x v_wmma_f32_16x16x32_bf16 per K-step,
// reusing its A fragment across the 4 N-subtiles. f32->bf16 uses the
// native hardware conversion (v_cvt_pk_bf16_f32).
// Requires T % 128 == 0, K % 32 == 0 (true for all call sites).
// =====================================================================
__global__ void gemm_wmma_bf16(const float* __restrict__ A,
                               const float* __restrict__ W,
                               const float* __restrict__ bias,
                               float* __restrict__ C,
                               int T, int K, int Nout, int tilesN4)
{
    __shared__ float smem[2 * 64 * 32];            // 2 phases x 64 cols x 32 K  (16 KB)

    const int lane  = threadIdx.x & 31;
    const int wave  = threadIdx.x >> 5;
    const int tn    = (blockIdx.x % tilesN4) * 64;
    const int tmw   = (blockIdx.x / tilesN4) * 128 + wave * 16;

    // ---- async stage of one B strip into LDS phase buffer ----
    auto stage = [&](int k0, int phase) {
#pragma unroll
        for (int c = 0; c < 2; ++c) {
            int chunk = threadIdx.x + c * 256;      // 0..511 : 16B chunks of the 8KB strip
            int col   = chunk >> 3;                 // 0..63
            int kp    = chunk & 7;                  // 16B unit within 128B row
            int rowW  = tn + col; if (rowW >= Nout) rowW = Nout - 1;   // clamp (junk cols never stored)
            const float* g = W + (size_t)rowW * K + k0 + kp * 4;
            lds_f* dstp = (lds_f*)&smem[phase * 2048 + chunk * 4];     // escapes smem; 32-bit LDS addr
            asm volatile("global_load_async_to_lds_b128 %0, %1, off"
                         :: "v"(dstp), "v"(g) : "memory");
        }
    };

    const int aRow = tmw + (lane & 15);             // lanes 0-15 & 16-31 hold same rows, different K halves
    const int aoff = (lane < 16) ? 0 : 8;           // A: K 0..7 & 16..23  /  8..15 & 24..31
    const int boff = (lane < 16) ? 0 : 8;           // B: K 0..7 & 16..23  /  8..15 & 24..31
    const float* __restrict__ arow = A + (size_t)aRow * K;

    v8f acc[4] = {};
    stage(0, 0);                                    // prologue

    for (int k0 = 0; k0 < K; k0 += 32) {
        const int phase = (k0 >> 5) & 1;
        if (k0 + 32 < K) {
            stage(k0 + 32, phase ^ 1);              // overlap next strip copy
            asm volatile("s_wait_asynccnt 0x2" ::: "memory");   // our 2 older copies done
        } else {
            asm volatile("s_wait_asynccnt 0x0" ::: "memory");
        }
        __syncthreads();                            // everyone's chunks visible

        // A fragment: 16 bf16 from 4 x b128 global loads + hw cvt
        BF16Frag a;
        {
            const v4f a0 = *(const v4f*)(arow + k0 + aoff);
            const v4f a1 = *(const v4f*)(arow + k0 + aoff + 4);
            const v4f a2 = *(const v4f*)(arow + k0 + 16 + aoff);
            const v4f a3 = *(const v4f*)(arow + k0 + 16 + aoff + 4);
#pragma unroll
            for (int i = 0; i < 4; ++i) {
                a.e[i]      = (__bf16)a0[i];
                a.e[4 + i]  = (__bf16)a1[i];
                a.e[8 + i]  = (__bf16)a2[i];
                a.e[12 + i] = (__bf16)a3[i];
            }
        }

        // 4 N-subtiles reuse the A fragment
#pragma unroll
        for (int sub = 0; sub < 4; ++sub) {
            const int colL = sub * 16 + (lane & 15);          // column within the 64-col strip
            const float* bp = smem + phase * 2048 + colL * 32 + boff;
            BF16Frag b;
            const v4f b0 = *(const v4f*)(bp);
            const v4f b1 = *(const v4f*)(bp + 4);
            const v4f b2 = *(const v4f*)(bp + 16);
            const v4f b3 = *(const v4f*)(bp + 20);
#pragma unroll
            for (int i = 0; i < 4; ++i) {
                b.e[i]      = (__bf16)b0[i];
                b.e[4 + i]  = (__bf16)b1[i];
                b.e[8 + i]  = (__bf16)b2[i];
                b.e[12 + i] = (__bf16)b3[i];
            }
            acc[sub] = __builtin_amdgcn_wmma_f32_16x16x32_bf16(
                           false, a.v, false, b.v, (short)0, acc[sub], false, false);
        }
        __syncthreads();                            // reads done before this phase is overwritten
    }

    // ---- store: C/D layout -> VGPR r holds M = r (lanes<16) / r+8 (lanes>=16) ----
    const int rbase = tmw + ((lane >= 16) ? 8 : 0);
#pragma unroll
    for (int sub = 0; sub < 4; ++sub) {
        const int colC = tn + sub * 16 + (lane & 15);
        if (colC < Nout) {
            const float bv = bias ? bias[colC] : 0.f;
#pragma unroll
            for (int r = 0; r < 8; ++r)
                C[(size_t)(rbase + r) * Nout + colC] = acc[sub][r] + bv;
        }
    }
}

// =====================================================================
// Per-(b,m) mean / unbiased std over L
// =====================================================================
__global__ void stats_kernel(const float* __restrict__ x, float* mean, float* stdv)
{
    __shared__ float ss[256], sq[256];
    const int bm = blockIdx.x, b = bm / M_, m = bm % M_;
    float s = 0.f, q = 0.f;
    for (int l = threadIdx.x; l < L_; l += 256) {
        float v = x[((size_t)b * L_ + l) * M_ + m];
        s += v; q += v * v;
    }
    ss[threadIdx.x] = s; sq[threadIdx.x] = q;
    __syncthreads();
    for (int st = 128; st > 0; st >>= 1) {
        if (threadIdx.x < st) { ss[threadIdx.x] += ss[threadIdx.x + st]; sq[threadIdx.x] += sq[threadIdx.x + st]; }
        __syncthreads();
    }
    if (threadIdx.x == 0) {
        float mu  = ss[0] / (float)L_;
        float var = (sq[0] - (float)L_ * mu * mu) / (float)(L_ - 1);
        mean[bm] = mu;
        stdv[bm] = sqrtf(fmaxf(var, 0.f)) + 1e-5f;
    }
}

// h[bm,n,d] = sum_p ((x[b, n*S+p, m]-mu)/sd) * pw[d,p] + pb[d] + pos[n,d]
__global__ void patch_embed_kernel(const float* __restrict__ x, const float* mean, const float* stdv,
                                   const float* __restrict__ pw, const float* __restrict__ pb,
                                   const float* __restrict__ pos, float* __restrict__ h, int n)
{
    int i = blockIdx.x * blockDim.x + threadIdx.x;
    if (i >= n) return;
    int d = i % D_; int r = i / D_; int nn = r % N_; int bm = r / N_;
    int b = bm / M_, m = bm % M_;
    float mu = mean[bm], inv = 1.f / stdv[bm];
    float acc = pb[d];
#pragma unroll
    for (int p = 0; p < P_; ++p) {
        float xv = x[((size_t)b * L_ + nn * S_ + p) * M_ + m];
        acc += (xv - mu) * inv * pw[d * P_ + p];
    }
    h[i] = acc + pos[nn * D_ + d];
}

// =====================================================================
// Dense mixing: out = sum softmax(rowa)[j]*la[j] + sum softmax(rowb)[j]*lb[j]
// =====================================================================
struct Ptr5 { const float* p[5]; };

__global__ void mix_kernel(Ptr5 la, int ca, const float* __restrict__ rowa,
                           Ptr5 lb, int cb, const float* __restrict__ rowb,
                           float* __restrict__ out, int n)
{
    int i = blockIdx.x * blockDim.x + threadIdx.x;
    if (i >= n) return;
    float w[5], acc = 0.f;
    {   float mx = -1e30f;
        for (int j = 0; j < ca; ++j) { w[j] = rowa[j]; mx = fmaxf(mx, w[j]); }
        float s = 0.f;
        for (int j = 0; j < ca; ++j) { w[j] = __expf(w[j] - mx); s += w[j]; }
        float inv = 1.f / s;
        for (int j = 0; j < ca; ++j) acc += w[j] * inv * la.p[j][i];
    }
    if (cb > 0) {
        float mx = -1e30f;
        for (int j = 0; j < cb; ++j) { w[j] = rowb[j]; mx = fmaxf(mx, w[j]); }
        float s = 0.f;
        for (int j = 0; j < cb; ++j) { w[j] = __expf(w[j] - mx); s += w[j]; }
        float inv = 1.f / s;
        for (int j = 0; j < cb; ++j) acc += w[j] * inv * lb.p[j][i];
    }
    out[i] = acc;
}

// LayerNorm over contiguous last dim D (thread per row)
__global__ void ln_d_kernel(const float* __restrict__ in, const float* __restrict__ w,
                            const float* __restrict__ b, float* __restrict__ out, int rows)
{
    int r = blockIdx.x * blockDim.x + threadIdx.x;
    if (r >= rows) return;
    const float* xp = in + (size_t)r * D_;
    float s = 0.f;
    for (int d = 0; d < D_; ++d) s += xp[d];
    float mu = s / (float)D_, q = 0.f;
    for (int d = 0; d < D_; ++d) { float dv = xp[d] - mu; q += dv * dv; }
    float rs = rsqrtf(q / (float)D_ + 1e-5f);
    float* op = out + (size_t)r * D_;
    for (int d = 0; d < D_; ++d) op[d] = (xp[d] - mu) * rs * w[d] + b[d];
}

// LayerNorm across N (stride D) for fixed (bm,d): buf layout (BM,N,D)
__global__ void ln_n_kernel(const float* __restrict__ in, const float* __restrict__ w,
                            const float* __restrict__ b, float* __restrict__ out, int n)
{
    int i = blockIdx.x * blockDim.x + threadIdx.x;
    if (i >= n) return;                         // n = BM_*D_
    int d = i % D_, bm = i / D_;
    const float* xp = in + (size_t)bm * N_ * D_ + d;
    float s = 0.f;
    for (int k = 0; k < N_; ++k) s += xp[(size_t)k * D_];
    float mu = s / (float)N_, q = 0.f;
    for (int k = 0; k < N_; ++k) { float dv = xp[(size_t)k * D_] - mu; q += dv * dv; }
    float rs = rsqrtf(q / (float)N_ + 1e-5f);
    float* op = out + (size_t)bm * N_ * D_ + d;
    for (int k = 0; k < N_; ++k) op[(size_t)k * D_] = (xp[(size_t)k * D_] - mu) * rs * w[k] + b[k];
}

// causal (rev=0) / anti-causal (rev=1, flipped taps) depthwise conv + SiLU
__global__ void conv_silu_kernel(const float* __restrict__ xz, const float* __restrict__ cw,
                                 const float* __restrict__ cb, float* __restrict__ u,
                                 int nseq, int slen, int rev, int n)
{
    int i = blockIdx.x * blockDim.x + threadIdx.x;
    if (i >= n) return;                        // n = nseq*slen*DI
    int c = i % DI_; int t = i / DI_;
    int pos = t % slen, seq = t / slen;
    float acc = cb[c];
#pragma unroll
    for (int dd = 0; dd < DC_; ++dd) {
        int p = rev ? pos + dd : pos - dd;
        if (p >= 0 && p < slen)
            acc += cw[c * DC_ + (DC_ - 1 - dd)] * xz[((size_t)(seq * slen + p)) * (2 * DI_) + c];
    }
    u[i] = siluf(acc);
}

// dt = softplus(xdbl[:, :8] @ dtw^T + dtb)
__global__ void dt_kernel(const float* __restrict__ xdbl, const float* __restrict__ dtw,
                          const float* __restrict__ dtb, float* __restrict__ dt, int n)
{
    int i = blockIdx.x * blockDim.x + threadIdx.x;
    if (i >= n) return;                        // n = T*DI
    int c = i % DI_; int t = i / DI_;
    float acc = dtb[c];
#pragma unroll
    for (int r = 0; r < DTR_; ++r) acc += xdbl[(size_t)t * XP_ + r] * dtw[c * DTR_ + r];
    dt[i] = softplusf(acc);
}

// selective scan, thread per (seq, channel); 16-state recurrence in registers
__global__ void scan_kernel(const float* __restrict__ dt, const float* __restrict__ xdbl,
                            const float* __restrict__ u, const float* __restrict__ xz,
                            const float* __restrict__ a_log, const float* __restrict__ dssm,
                            float* __restrict__ y, int nseq, int slen, int rev, int n)
{
    int i = blockIdx.x * blockDim.x + threadIdx.x;
    if (i >= n) return;                        // n = nseq*DI
    int c = i % DI_; int seq = i / DI_;
    float A[NST_], hh[NST_];
#pragma unroll
    for (int k = 0; k < NST_; ++k) { A[k] = -__expf(a_log[c * NST_ + k]); hh[k] = 0.f; }
    const float Dv = dssm[c];
    for (int st = 0; st < slen; ++st) {
        int pos = rev ? (slen - 1 - st) : st;
        size_t t = (size_t)seq * slen + pos;
        float dtv = dt[t * DI_ + c];
        float uv  = u[t * DI_ + c];
        float du  = dtv * uv;
        float yv  = 0.f;
#pragma unroll
        for (int k = 0; k < NST_; ++k) {
            float Bv = xdbl[t * XP_ + DTR_ + k];
            float Cv = xdbl[t * XP_ + DTR_ + NST_ + k];
            hh[k] = hh[k] * __expf(dtv * A[k]) + du * Bv;
            yv += hh[k] * Cv;
        }
        yv += uv * Dv;
        float zv = xz[t * (2 * DI_) + DI_ + c];
        y[t * DI_ + c] = yv * siluf(zv);
    }
}

__global__ void add2_kernel(const float* a, const float* b, float* out, int n)
{
    int i = blockIdx.x * blockDim.x + threadIdx.x;
    if (i < n) out[i] = a[i] + b[i];
}

// (B,M,N,D) -> (B,N,M,D)
__global__ void t_bmnd_bnmd_kernel(const float* __restrict__ in, float* __restrict__ out, int n)
{
    int i = blockIdx.x * blockDim.x + threadIdx.x;
    if (i >= n) return;
    int d = i % D_; int r = i / D_; int nn = r % N_; r /= N_; int m = r % M_; int b = r / M_;
    out[(((size_t)b * N_ + nn) * M_ + m) * D_ + d] = in[i];
}

// ch_out(BMND) = fwd(BNMD) + bwd(BNMD) + ch_in(BMND)
__global__ void combine_ch_kernel(const float* __restrict__ fwd, const float* __restrict__ bwd,
                                  const float* __restrict__ mixc, float* __restrict__ out, int n)
{
    int i = blockIdx.x * blockDim.x + threadIdx.x;
    if (i >= n) return;
    int d = i % D_; int r = i / D_; int nn = r % N_; r /= N_; int m = r % M_; int b = r / M_;
    size_t j = (((size_t)b * N_ + nn) * M_ + m) * D_ + d;
    out[i] = fwd[j] + bwd[j] + mixc[i];
}

// out[b,pred,m] = head[(b*M+m)*PRED+pred]*std[bm] + mean[bm]
__global__ void final_kernel(const float* __restrict__ ho, const float* mean, const float* stdv,
                             float* __restrict__ out, int n)
{
    int i = blockIdx.x * blockDim.x + threadIdx.x;
    if (i >= n) return;
    int m = i % M_; int r = i / M_; int pr = r % PRED_; int b = r / PRED_;
    int bm = b * M_ + m;
    out[i] = ho[(size_t)bm * PRED_ + pr] * stdv[bm] + mean[bm];
}

// =====================================================================
// Host orchestration
// =====================================================================
static inline int cdiv(int a, int b) { return (a + b - 1) / b; }

extern "C" void kernel_launch(void* const* d_in, const int* in_sizes, int n_in,
                              void* d_out, int out_size, void* d_ws, size_t ws_size,
                              hipStream_t stream)
{
    const float* x        = (const float*)d_in[0];
    const float* patch_w  = (const float*)d_in[1];
    const float* patch_b  = (const float*)d_in[2];
    const float* pos      = (const float*)d_in[3];
    const float* alpha    = (const float*)d_in[4];
    const float* beta     = (const float*)d_in[5];
    const float* theta    = (const float*)d_in[6];
    const float* gammaw   = (const float*)d_in[7];
    const float* tnw      = (const float*)d_in[8];
    const float* tnb      = (const float*)d_in[9];
    const float* cnw      = (const float*)d_in[10];
    const float* cnb      = (const float*)d_in[11];
    const float* in_proj  = (const float*)d_in[12];
    const float* conv_w   = (const float*)d_in[13];
    const float* conv_b   = (const float*)d_in[14];
    const float* x_proj   = (const float*)d_in[15];
    const float* dt_w     = (const float*)d_in[16];
    const float* dt_b     = (const float*)d_in[17];
    const float* A_log    = (const float*)d_in[18];
    const float* D_ssm    = (const float*)d_in[19];
    const float* out_proj = (const float*)d_in[20];
    const float* n2w1     = (const float*)d_in[21];
    const float* n2b1     = (const float*)d_in[22];
    const float* n2w2     = (const float*)d_in[23];
    const float* n2b2     = (const float*)d_in[24];
    const float* head_w   = (const float*)d_in[25];
    const float* head_b   = (const float*)d_in[26];
    float* out = (float*)d_out;

    // ---- workspace carve-out (~440 MB f32) ----
    size_t off = 0;
    auto alloc = [&](size_t nf) { float* p = (float*)d_ws + off; off += nf; return p; };
    const size_t HN = (size_t)BM_ * N_ * D_;      // 4,128,768  (also == TTOK_*D_)
    float* mean = alloc(BM_);
    float* stdv = alloc(BM_);
    float* h    = alloc(HN);
    float* yT[NL_ + 1]; float* yC[NL_ + 1];
    yT[0] = h; yC[0] = h;
    for (int l = 1; l <= NL_; ++l) { yT[l] = alloc(HN); yC[l] = alloc(HN); }
    float* mixT  = alloc(HN);
    float* lnT   = alloc(HN);
    float* mixC  = alloc(HN);
    float* chrT  = alloc(HN);
    float* lnC   = alloc(HN);
    float* projF = alloc(HN);
    float* projB = alloc(HN);
    float* xz    = alloc((size_t)TTOK_ * 2 * DI_);
    float* ubuf  = alloc((size_t)TTOK_ * DI_);
    float* xdbl  = alloc((size_t)TTOK_ * XP_);
    float* dtbuf = alloc((size_t)TTOK_ * DI_);
    float* yss   = alloc((size_t)TTOK_ * DI_);
    float* headout = alloc((size_t)BM_ * PRED_);

    auto gemm = [&](const float* A, const float* W, const float* bias, float* C,
                    int T, int K, int Nout) {
        int tilesN4 = cdiv(Nout, 64);
        int blocks  = (T / 128) * tilesN4;        // exact: all T are multiples of 128
        gemm_wmma_bf16<<<blocks, 256, 0, stream>>>(A, W, bias, C, T, K, Nout, tilesN4);
    };

    auto run_mamba = [&](const float* inp, float* outbuf, int l, int role,
                         int nseq, int slen, int rev) {
        const int T = nseq * slen;
        const size_t lr = (size_t)l * 3 + role;
        const float* iw  = in_proj  + lr * (2 * DI_) * D_;
        const float* cw  = conv_w   + lr * DI_ * DC_;
        const float* cb  = conv_b   + lr * DI_;
        const float* xw  = x_proj   + lr * XP_ * DI_;
        const float* dtw = dt_w     + lr * DI_ * DTR_;
        const float* dtb = dt_b     + lr * DI_;
        const float* al  = A_log    + lr * DI_ * NST_;
        const float* dp  = D_ssm    + lr * DI_;
        const float* ow  = out_proj + lr * D_ * DI_;

        gemm(inp, iw, nullptr, xz, T, D_, 2 * DI_);                 // xz = inp @ in_w^T
        int nconv = T * DI_;
        conv_silu_kernel<<<cdiv(nconv, 256), 256, 0, stream>>>(xz, cw, cb, ubuf, nseq, slen, rev, nconv);
        gemm(ubuf, xw, nullptr, xdbl, T, DI_, XP_);                 // xdbl = u @ x_proj^T
        dt_kernel<<<cdiv(nconv, 256), 256, 0, stream>>>(xdbl, dtw, dtb, dtbuf, nconv);
        int nscan = nseq * DI_;
        scan_kernel<<<cdiv(nscan, 256), 256, 0, stream>>>(dtbuf, xdbl, ubuf, xz, al, dp, yss,
                                                          nseq, slen, rev, nscan);
        gemm(yss, ow, nullptr, outbuf, T, DI_, D_);                 // out = y @ out_proj^T
    };

    const int nH = (int)HN;

    stats_kernel<<<BM_, 256, 0, stream>>>(x, mean, stdv);
    patch_embed_kernel<<<cdiv(nH, 256), 256, 0, stream>>>(x, mean, stdv, patch_w, patch_b, pos, h, nH);

    for (int l = 0; l < NL_; ++l) {
        // -------- token path --------
        Ptr5 lt = {}, lc = {};
        for (int i = 0; i <= l; ++i) { lt.p[i] = yT[i]; lc.p[i] = yC[i]; }
        mix_kernel<<<cdiv(nH, 256), 256, 0, stream>>>(lt, l + 1, alpha + l * NL_,
                                                      lc, l + 1, beta + l * NL_, mixT, nH);
        ln_d_kernel<<<cdiv(TTOK_, 256), 256, 0, stream>>>(mixT, tnw + l * D_, tnb + l * D_, lnT, TTOK_);
        run_mamba(lnT, projF, l, 0, BM_, N_, 0);
        add2_kernel<<<cdiv(nH, 256), 256, 0, stream>>>(projF, mixT, yT[l + 1], nH);

        // -------- channel path --------
        Ptr5 ltt = {}, lg = {};
        for (int i = 0; i <= l + 1; ++i) ltt.p[i] = yT[i];
        for (int i = 0; i <= l; ++i)     lg.p[i]  = yC[i];
        mix_kernel<<<cdiv(nH, 256), 256, 0, stream>>>(ltt, l + 2, theta + l * (NL_ + 1),
                                                      lg, l + 1, gammaw + l * NL_, mixC, nH);
        t_bmnd_bnmd_kernel<<<cdiv(nH, 256), 256, 0, stream>>>(mixC, chrT, nH);
        ln_d_kernel<<<cdiv(TTOK_, 256), 256, 0, stream>>>(chrT, cnw + l * D_, cnb + l * D_, lnC, TTOK_);
        run_mamba(lnC, projF, l, 1, B_ * N_, M_, 0);   // forward
        run_mamba(lnC, projB, l, 2, B_ * N_, M_, 1);   // backward (flip == anti-causal + reverse scan)
        combine_ch_kernel<<<cdiv(nH, 256), 256, 0, stream>>>(projF, projB, mixC, yC[l + 1], nH);
    }

    // ---- epilogue: Norm2D + head + denorm ----
    ln_d_kernel<<<cdiv(TTOK_, 256), 256, 0, stream>>>(yC[NL_], n2w1, n2b1, mixT, TTOK_);
    int nND = BM_ * D_;
    ln_n_kernel<<<cdiv(nND, 256), 256, 0, stream>>>(mixT, n2w2, n2b2, lnT, nND);
    gemm(lnT, head_w, head_b, headout, BM_, N_ * D_, PRED_);   // K = 8064, Nout = 96
    int nOut = B_ * PRED_ * M_;
    final_kernel<<<cdiv(nOut, 256), 256, 0, stream>>>(headout, mean, stdv, out, nOut);
}
